// FlareMPNNLSTM_22522808500499
// MI455X (gfx1250) — matrix operand order repaired and checked by reading, not compiled, not run on hardware
//
#include <hip/hip_runtime.h>
#include <hip/hip_bf16.h>

// ---------------------------------------------------------------------------
// Types / helpers
// ---------------------------------------------------------------------------
typedef __attribute__((ext_vector_type(16))) __bf16 v16bf;
typedef __attribute__((ext_vector_type(8)))  float  v8f;

static __device__ __forceinline__ unsigned short f32_to_bf16_raw(float f) {
  union { float f; unsigned u; } v; v.f = f;
  unsigned r = v.u + 0x7FFFu + ((v.u >> 16) & 1u);   // round-to-nearest-even
  return (unsigned short)(r >> 16);
}
static __device__ __forceinline__ unsigned pack2bf(float a, float b) {
  return (unsigned)f32_to_bf16_raw(a) | ((unsigned)f32_to_bf16_raw(b) << 16);
}
static __device__ __forceinline__ __bf16 bfbits(unsigned short u) {
  return __builtin_bit_cast(__bf16, u);
}
static __device__ __forceinline__ float sigmoidf_(float x) {
  return 1.0f / (1.0f + __expf(-x));
}

// A-fragment (16x32 bf16) per CDNA5 layout: lane L -> M=L%16, half=L/16;
// VGPR v<4 -> K = 2v + 8*half ; v>=4 -> K = 16 + 2(v-4) + 8*half
// For fixed lane this is two contiguous 8-element (16B) runs -> ds_load_b128 x2
static __device__ __forceinline__ v16bf load_a_frag(const unsigned short* sA,
                                                    int ldk, int m, int half, int k0) {
  v16bf a;
#pragma unroll
  for (int v = 0; v < 8; ++v) {
    int kk = k0 + ((v & 4) ? 16 : 0) + (v & 3) * 2 + half * 8;
    a[2 * v]     = bfbits(sA[m * ldk + kk]);
    a[2 * v + 1] = bfbits(sA[m * ldk + kk + 1]);
  }
  return a;
}

// ---------------------------------------------------------------------------
// Utility kernels
// ---------------------------------------------------------------------------
__global__ void k_zero4(float4* __restrict__ p, long n4) {
  long i = (long)blockIdx.x * blockDim.x + threadIdx.x;
  if (i < n4) p[i] = make_float4(0.f, 0.f, 0.f, 0.f);
}

__global__ void k_deg(const int* __restrict__ dst, float* __restrict__ deg, int E) {
  int e = blockIdx.x * blockDim.x + threadIdx.x;
  if (e < E) atomicAdd(&deg[dst[e]], 1.0f);
}

__global__ void k_norm(const int* __restrict__ dst, const float* __restrict__ deg,
                       float* __restrict__ norm, int E) {
  int e = blockIdx.x * blockDim.x + threadIdx.x;
  if (e < E) norm[e] = 1.0f / fmaxf(deg[dst[e]], 1.0f);
}

// gather(src)*norm -> scatter_add(dst); one thread per (edge, 4-float chunk)
__global__ void k_mp(const float* __restrict__ hin, const int* __restrict__ src,
                     const int* __restrict__ dst, const float* __restrict__ norm,
                     float* __restrict__ out, int E, int sh /* log2(F/4) */) {
  size_t i = (size_t)blockIdx.x * blockDim.x + threadIdx.x;
  size_t total = (size_t)E << sh;
  if (i >= total) return;
  int e  = (int)(i >> sh);
  int ch = (int)(i & ((1u << sh) - 1u));
  int F  = 4 << sh;
  float nw = norm[e];
  const float4* s = (const float4*)(hin + (size_t)src[e] * F) + ch;
  float4 v = *s;
  float* o = out + (size_t)dst[e] * F + ch * 4;
  atomicAdd(o + 0, v.x * nw);
  atomicAdd(o + 1, v.y * nw);
  atomicAdd(o + 2, v.z * nw);
  atomicAdd(o + 3, v.w * nw);
}

// ---------------------------------------------------------------------------
// Pack fp32 weights W[K, Ncols] into fragment-major bf16:
//   Wp[((tk*(Ncols/16) + tn)*32 + lane)*16 + i]  -- 32B contiguous per lane
// so a B fragment is a single aligned 32-byte (v16bf) load.
// ---------------------------------------------------------------------------
__global__ void k_pack_b(const float* __restrict__ W, unsigned short* __restrict__ Wp,
                         int K, int Ncols) {
  int idx = blockIdx.x * blockDim.x + threadIdx.x;     // one thread = one lane-frag
  int total = (K / 32) * (Ncols / 16) * 32;
  if (idx >= total) return;
  int lane = idx & 31;
  int t    = idx >> 5;
  int ntn  = Ncols / 16;
  int tn   = t % ntn;
  int tk   = t / ntn;
  int n    = tn * 16 + (lane & 15);
  int half = lane >> 4;
  unsigned short* o = Wp + (size_t)idx * 16;
#pragma unroll
  for (int v = 0; v < 8; ++v) {
    int kk = tk * 32 + ((v & 4) ? 16 : 0) + (v & 3) * 2 + half * 8;
    o[2 * v]     = f32_to_bf16_raw(W[(size_t)kk * Ncols + n]);
    o[2 * v + 1] = f32_to_bf16_raw(W[(size_t)(kk + 1) * Ncols + n]);
  }
}

// ---------------------------------------------------------------------------
// GCN GEMM: out[N,128] = (RELU?) A[N,K] @ W[K,128]   (bf16 WMMA, f32 accum)
// block = 256 threads (8 waves) -> 16 rows x 128 cols; wave w -> col tile w
// ---------------------------------------------------------------------------
template <int RELU>
__global__ void k_gemm_gcn(const float* __restrict__ A,
                           const unsigned short* __restrict__ Wp,
                           float* __restrict__ out, int N, int K) {
  __shared__ unsigned int sA[16 * 64];               // 16 rows x (K<=128) bf16
  const int row0 = blockIdx.x * 16;
  const int tid  = threadIdx.x;
  const int K2   = K >> 1;
  const bool full = (row0 + 16 <= N);

  if (full) {
    for (int idx = tid; idx < 16 * K2; idx += 256) {
      int r = idx / K2, k2 = idx - r * K2;
      float2 v = *(const float2*)(A + (size_t)(row0 + r) * K + 2 * k2);
      sA[idx] = pack2bf(v.x, v.y);
    }
  } else {
    for (int idx = tid; idx < 16 * K2; idx += 256) {
      int r = idx / K2, k2 = idx - r * K2;
      float2 v = {0.f, 0.f};
      if (row0 + r < N) v = *(const float2*)(A + (size_t)(row0 + r) * K + 2 * k2);
      sA[idx] = pack2bf(v.x, v.y);
    }
  }
  __syncthreads();

  const int wave = tid >> 5, lane = tid & 31;
  const int half = lane >> 4, l16 = lane & 15;
  const int col0 = wave * 16;
  const int ntn  = 8;                                // 128/16 col tiles

  v8f acc = {};
  for (int k0 = 0; k0 < K; k0 += 32) {
    int tk = k0 >> 5;
    const unsigned short* fb =
        Wp + ((((size_t)tk * ntn + wave) * 32 + lane) << 4);
    __builtin_prefetch(fb + (size_t)ntn * 512, 0, 3);   // next tk, same tn
    v16bf a = load_a_frag((const unsigned short*)sA, K, l16, half, k0);
    v16bf b = *(const v16bf*)fb;                        // 32B aligned vector load
    acc = __builtin_amdgcn_wmma_f32_16x16x32_bf16(false, a, false, b,
                                                  (short)0, acc, false, false);
  }

  if (full) {
#pragma unroll
    for (int v = 0; v < 8; ++v) {
      float val = acc[v];
      if (RELU) val = fmaxf(val, 0.0f);
      out[(size_t)(row0 + v + 8 * half) * 128 + col0 + l16] = val;
    }
  } else {
#pragma unroll
    for (int v = 0; v < 8; ++v) {
      int r = row0 + v + 8 * half;
      if (r < N) {
        float val = acc[v];
        if (RELU) val = fmaxf(val, 0.0f);
        out[(size_t)r * 128 + col0 + l16] = val;
      }
    }
  }
}

// ---------------------------------------------------------------------------
// Fused LSTM cell: z = [g | h] @ W[256,512] + bias ; gates ; h,c in place
// (each block owns its 16 rows exclusively).
// block = 256 threads (8 waves) -> 16 rows x 512 cols.
// Each wave holds 4 accumulators (cols [64w,64w+64)) so one A-fragment load
// feeds 4 WMMAs per K-step.
// ---------------------------------------------------------------------------
__global__ void k_lstm(const float* __restrict__ g,
                       float* __restrict__ h, float* __restrict__ c,
                       const unsigned short* __restrict__ Wp,
                       const float* __restrict__ bias, int N) {
  __shared__ unsigned int sA[16 * 128];              // 8 KB  (bf16 [g|h] tile)
  __shared__ float sZ[16 * 512];                     // 32 KB (f32 z tile)
  const int row0 = blockIdx.x * 16;
  const int tid  = threadIdx.x;
  const bool full = (row0 + 16 <= N);

  for (int idx = tid; idx < 16 * 128; idx += 256) {  // pairs: 256 k per row
    int r = idx >> 7, k2 = idx & 127;
    float2 v = {0.f, 0.f};
    if (full || row0 + r < N) {
      int k = k2 * 2;
      const float* sp = (k < 128) ? (g + (size_t)(row0 + r) * 128 + k)
                                  : (h + (size_t)(row0 + r) * 128 + (k - 128));
      v = *(const float2*)sp;
    }
    sA[idx] = pack2bf(v.x, v.y);
  }
  __syncthreads();

  const int wave = tid >> 5, lane = tid & 31;
  const int half = lane >> 4, l16 = lane & 15;
  const int ntn  = 32;                               // 512/16 col tiles
  const int tn0  = wave * 4;

  v8f acc0 = {}, acc1 = {}, acc2 = {}, acc3 = {};
  for (int k0 = 0; k0 < 256; k0 += 32) {
    int tk = k0 >> 5;
    const unsigned short* fb =
        Wp + ((((size_t)tk * ntn + tn0) * 32 + lane) << 4);
    __builtin_prefetch(fb + (size_t)ntn * 512, 0, 3);         // next tk
    __builtin_prefetch(fb + (size_t)ntn * 512 + 1024, 0, 3);
    v16bf a = load_a_frag((const unsigned short*)sA, 256, l16, half, k0);
    v16bf b0 = *(const v16bf*)(fb);
    v16bf b1 = *(const v16bf*)(fb + 512);        // tn0+1 (32 lanes * 16 bf16)
    v16bf b2 = *(const v16bf*)(fb + 1024);       // tn0+2
    v16bf b3 = *(const v16bf*)(fb + 1536);       // tn0+3
    acc0 = __builtin_amdgcn_wmma_f32_16x16x32_bf16(false, a, false, b0,
                                                   (short)0, acc0, false, false);
    acc1 = __builtin_amdgcn_wmma_f32_16x16x32_bf16(false, a, false, b1,
                                                   (short)0, acc1, false, false);
    acc2 = __builtin_amdgcn_wmma_f32_16x16x32_bf16(false, a, false, b2,
                                                   (short)0, acc2, false, false);
    acc3 = __builtin_amdgcn_wmma_f32_16x16x32_bf16(false, a, false, b3,
                                                   (short)0, acc3, false, false);
  }
#pragma unroll
  for (int v = 0; v < 8; ++v) {
    int zr = (v + 8 * half) * 512 + l16;
    sZ[zr + (tn0 + 0) * 16] = acc0[v];
    sZ[zr + (tn0 + 1) * 16] = acc1[v];
    sZ[zr + (tn0 + 2) * 16] = acc2[v];
    sZ[zr + (tn0 + 3) * 16] = acc3[v];
  }
  __syncthreads();

  // gating: 16 rows x 128 hidden, split i|f|g|o along columns of z
  for (int idx = tid; idx < 16 * 128; idx += 256) {
    int r = idx >> 7, j = idx & 127;
    int n = row0 + r;
    if (full || n < N) {
      float zi = sZ[r * 512 + j]       + bias[j];
      float zf = sZ[r * 512 + 128 + j] + bias[128 + j];
      float zg = sZ[r * 512 + 256 + j] + bias[256 + j];
      float zo = sZ[r * 512 + 384 + j] + bias[384 + j];
      size_t p = (size_t)n * 128 + j;
      float cn = sigmoidf_(zf) * c[p] + sigmoidf_(zi) * tanhf(zg);
      c[p] = cn;
      h[p] = sigmoidf_(zo) * tanhf(cn);
    }
  }
}

// ---------------------------------------------------------------------------
// Output projection: y[n] = h2[n,:] . outW + outb   (one wave32 per row)
// ---------------------------------------------------------------------------
__global__ void k_out(const float* __restrict__ h2, const float* __restrict__ outW,
                      const float* __restrict__ outb, float* __restrict__ y, int N) {
  int row  = blockIdx.x * 8 + (threadIdx.x >> 5);
  int lane = threadIdx.x & 31;
  if (row >= N) return;
  float s = 0.0f;
#pragma unroll
  for (int j = 0; j < 4; ++j)
    s += h2[(size_t)row * 128 + j * 32 + lane] * outW[j * 32 + lane];
#pragma unroll
  for (int off = 16; off > 0; off >>= 1) s += __shfl_xor(s, off, 32);
  if (lane == 0) y[row] = s + outb[0];
}

// ---------------------------------------------------------------------------
// Host launch
// ---------------------------------------------------------------------------
extern "C" void kernel_launch(void* const* d_in, const int* in_sizes, int n_in,
                              void* d_out, int out_size, void* d_ws, size_t ws_size,
                              hipStream_t stream) {
  const float* x   = (const float*)d_in[0];
  const int*   src = (const int*)  d_in[1];
  const int*   dst = (const int*)  d_in[2];
  const float* W1  = (const float*)d_in[3];
  const float* W2  = (const float*)d_in[4];
  const float* L1W = (const float*)d_in[5];
  const float* L1b = (const float*)d_in[6];
  const float* L2W = (const float*)d_in[7];
  const float* L2b = (const float*)d_in[8];
  const float* oW  = (const float*)d_in[9];
  const float* ob  = (const float*)d_in[10];

  const int E = in_sizes[1];
  const int H = in_sizes[6] / 4;       // 128
  const int F = in_sizes[3] / H;       // 64
  const int T = 4;
  const int N = out_size / T;          // 100000

  // ---- workspace carve-out ------------------------------------------------
  char*  ws  = (char*)d_ws;
  size_t off = 0;
  auto alloc = [&](size_t bytes) -> void* {
    void* p = ws + off;
    off += (bytes + 255) & ~(size_t)255;
    return p;
  };
  float* deg  = (float*)alloc((size_t)N * 4);
  float* nrm  = (float*)alloc((size_t)E * 4);
  float* mpA  = (float*)alloc((size_t)N * F * 4);
  float* g1   = (float*)alloc((size_t)N * H * 4);
  float* mpB  = (float*)alloc((size_t)N * H * 4);
  float* g2   = (float*)alloc((size_t)N * H * 4);
  float* h1   = (float*)alloc((size_t)N * H * 4);   // h1,c1,h2,c2 contiguous
  float* c1   = (float*)alloc((size_t)N * H * 4);
  float* h2   = (float*)alloc((size_t)N * H * 4);
  float* c2   = (float*)alloc((size_t)N * H * 4);
  unsigned short* W1p = (unsigned short*)alloc((size_t)F * H * 2);
  unsigned short* W2p = (unsigned short*)alloc((size_t)H * H * 2);
  unsigned short* L1p = (unsigned short*)alloc((size_t)2 * H * 4 * H * 2);
  unsigned short* L2p = (unsigned short*)alloc((size_t)2 * H * 4 * H * 2);
  (void)ws_size; (void)n_in; (void)c1; (void)c2;

  auto blocks = [](long n, int bs) { return (unsigned)((n + bs - 1) / bs); };

  // ---- one-time per launch: packed bf16 weights, degrees, norms, states ---
  k_pack_b<<<blocks((F / 32) * (H / 16) * 32, 256), 256, 0, stream>>>(W1, W1p, F, H);
  k_pack_b<<<blocks((H / 32) * (H / 16) * 32, 256), 256, 0, stream>>>(W2, W2p, H, H);
  k_pack_b<<<blocks((2 * H / 32) * (4 * H / 16) * 32, 256), 256, 0, stream>>>(
      L1W, L1p, 2 * H, 4 * H);
  k_pack_b<<<blocks((2 * H / 32) * (4 * H / 16) * 32, 256), 256, 0, stream>>>(
      L2W, L2p, 2 * H, 4 * H);

  k_zero4<<<blocks((long)N / 4, 256), 256, 0, stream>>>((float4*)deg, (long)N / 4);
  k_deg<<<blocks(E, 256), 256, 0, stream>>>(dst, deg, E);
  k_norm<<<blocks(E, 256), 256, 0, stream>>>(dst, deg, nrm, E);

  k_zero4<<<blocks((long)N * H, 256), 256, 0, stream>>>((float4*)h1, (long)N * H);
  // note: zeroes h1,c1,h2,c2 (4*N*H floats = N*H float4s)

  const int shF = __builtin_ctz(F / 4);   // 4
  const int shH = __builtin_ctz(H / 4);   // 5
  const unsigned gemmBlocks = blocks(N, 16);

  // ---- time loop ----------------------------------------------------------
  for (int t = 0; t < T; ++t) {
    const float* xt = x + (size_t)t * N * F;

    k_zero4<<<blocks((long)N * F / 4, 256), 256, 0, stream>>>((float4*)mpA,
                                                              (long)N * F / 4);
    k_mp<<<blocks((long)E << shF, 256), 256, 0, stream>>>(xt, src, dst, nrm,
                                                          mpA, E, shF);
    k_gemm_gcn<1><<<gemmBlocks, 256, 0, stream>>>(mpA, W1p, g1, N, F);

    k_zero4<<<blocks((long)N * H / 4, 256), 256, 0, stream>>>((float4*)mpB,
                                                              (long)N * H / 4);
    k_mp<<<blocks((long)E << shH, 256), 256, 0, stream>>>(g1, src, dst, nrm,
                                                          mpB, E, shH);
    k_gemm_gcn<0><<<gemmBlocks, 256, 0, stream>>>(mpB, W2p, g2, N, H);

    k_lstm<<<gemmBlocks, 256, 0, stream>>>(g2, h1, c1, L1p, L1b, N);
    k_lstm<<<gemmBlocks, 256, 0, stream>>>(h1, h2, c2, L2p, L2b, N);

    k_out<<<blocks(N, 8), 256, 0, stream>>>(h2, oW, ob,
                                            (float*)d_out + (size_t)t * N, N);
  }
}
